// ScaledRadialKernel_19387482374739
// MI455X (gfx1250) — matrix operand order repaired and checked by reading, not compiled, not run on hardware
//
#include <hip/hip_runtime.h>

// ---------------------------------------------------------------------------
// scores = (q q^T + k k^T - 2 q k^T)/sqrt(512)  ==  U V^T / sqrt(512)
//   U = [q | k]        (8192 x 1024)
//   V = [q - 2k | k]   (8192 x 1024)
// then row-wise softmax.  GEMM runs on v_wmma_f32_16x16x32_bf16.
// ---------------------------------------------------------------------------

typedef __bf16 bf16_t;
typedef bf16_t v16bf __attribute__((ext_vector_type(16)));
typedef bf16_t v8bf  __attribute__((ext_vector_type(8)));
typedef bf16_t v4bf  __attribute__((ext_vector_type(4)));
typedef float  v8f   __attribute__((ext_vector_type(8)));

#define N_TOK 8192
#define DDIM  512
#define LDSK  72   // 64 bf16 + 8 pad -> 144B row stride, conflict-free frags
#define SCALE 0.04419417382415922f  // 1/sqrt(512)

__global__ __launch_bounds__(128)
void fused_gram_gemm(const float* __restrict__ q,
                     const float* __restrict__ kin,
                     float* __restrict__ out) {
  __shared__ alignas(16) bf16_t As[128][LDSK];   // U tile  (rows of scores)
  __shared__ alignas(16) bf16_t Bs[128][LDSK];   // V tile  (cols of scores)

  const int t    = threadIdx.x;
  const int lane = t & 31;
  const int wave = t >> 5;        // 4 waves, 2x2 grid of 64x64 wave tiles
  const int lh   = lane & 15;
  const int hi   = lane >> 4;
  const int wr   = (wave >> 1) * 64;
  const int wc   = (wave & 1) * 64;

  const int row0 = blockIdx.y * 128;
  const int col0 = blockIdx.x * 128;

  // global->LDS loader mapping: 16 float4 columns x 8 rows, 16 iters
  const int lr = t >> 4;          // 0..7
  const int lc = t & 15;          // float4 column 0..15

  v8f acc[4][4] = {};

  for (int kb = 0; kb < 16; ++kb) {            // 16 K-blocks of 64 over K=1024
    // ---- stage A tile: U = (kb<8 ? q : k), convert f32 -> bf16 ----
    {
      const float* src = (kb < 8)
          ? (q   + (size_t)row0 * DDIM + kb * 64)
          : (kin + (size_t)row0 * DDIM + (kb - 8) * 64);
      #pragma unroll
      for (int i = 0; i < 16; ++i) {
        const int r = lr + 8 * i;
        const float4 v = *(const float4*)(src + (size_t)r * DDIM + lc * 4);
        v4bf p = { (bf16_t)v.x, (bf16_t)v.y, (bf16_t)v.z, (bf16_t)v.w };
        *(v4bf*)&As[r][lc * 4] = p;
      }
    }
    // ---- stage B tile: V = (kb<8 ? q-2k : k) fused during load ----
    if (kb < 8) {
      const float* sq = q   + (size_t)col0 * DDIM + kb * 64;
      const float* sk = kin + (size_t)col0 * DDIM + kb * 64;
      #pragma unroll
      for (int i = 0; i < 16; ++i) {
        const int r = lr + 8 * i;
        const float4 a = *(const float4*)(sq + (size_t)r * DDIM + lc * 4);
        const float4 b = *(const float4*)(sk + (size_t)r * DDIM + lc * 4);
        v4bf p = { (bf16_t)(a.x - 2.0f * b.x), (bf16_t)(a.y - 2.0f * b.y),
                   (bf16_t)(a.z - 2.0f * b.z), (bf16_t)(a.w - 2.0f * b.w) };
        *(v4bf*)&Bs[r][lc * 4] = p;
      }
    } else {
      const float* sk = kin + (size_t)col0 * DDIM + (kb - 8) * 64;
      #pragma unroll
      for (int i = 0; i < 16; ++i) {
        const int r = lr + 8 * i;
        const float4 v = *(const float4*)(sk + (size_t)r * DDIM + lc * 4);
        v4bf p = { (bf16_t)v.x, (bf16_t)v.y, (bf16_t)v.z, (bf16_t)v.w };
        *(v4bf*)&Bs[r][lc * 4] = p;
      }
    }
    __syncthreads();

    // ---- two K=32 WMMA steps over the 64-wide LDS stage ----
    #pragma unroll
    for (int ks = 0; ks < 64; ks += 32) {
      v16bf af[4], bfr[4];
      #pragma unroll
      for (int m = 0; m < 4; ++m) {
        // A frag: lane = row, 16 contiguous K (half selected by lane>=16)
        const bf16_t* ap = &As[wr + m * 16 + lh][ks + hi * 16];
        const v8bf lo = *(const v8bf*)ap;
        const v8bf hh = *(const v8bf*)(ap + 8);
        af[m] = __builtin_shufflevector(lo, hh, 0,1,2,3,4,5,6,7,
                                                8,9,10,11,12,13,14,15);
      }
      #pragma unroll
      for (int n = 0; n < 4; ++n) {
        // B frag: lane = column, 16 contiguous K (half selected by lane>=16)
        const bf16_t* bp = &Bs[wc + n * 16 + lh][ks + hi * 16];
        const v8bf lo = *(const v8bf*)bp;
        const v8bf hh = *(const v8bf*)(bp + 8);
        bfr[n] = __builtin_shufflevector(lo, hh, 0,1,2,3,4,5,6,7,
                                                 8,9,10,11,12,13,14,15);
      }
      #pragma unroll
      for (int m = 0; m < 4; ++m)
        #pragma unroll
        for (int n = 0; n < 4; ++n)
          acc[m][n] = __builtin_amdgcn_wmma_f32_16x16x32_bf16(
              false, af[m], false, bfr[n], (short)0, acc[m][n], false, false);
    }
    __syncthreads();
  }

  // ---- store scaled scores; D layout: VGPR r -> M=r(+8 for hi lanes), N=lane%16
  #pragma unroll
  for (int m = 0; m < 4; ++m) {
    #pragma unroll
    for (int n = 0; n < 4; ++n) {
      const int rbase = row0 + wr + m * 16 + hi * 8;
      const int cbase = col0 + wc + n * 16 + lh;
      #pragma unroll
      for (int r = 0; r < 8; ++r)
        out[(size_t)(rbase + r) * N_TOK + cbase] = acc[m][n][r] * SCALE;
    }
  }
}

// ---------------------------------------------------------------------------
// In-place numerically stable row softmax: 1 block / row, 32 f32 per thread.
// ---------------------------------------------------------------------------
__global__ __launch_bounds__(256)
void softmax_rows(float* __restrict__ out) {
  __shared__ float red[256];
  const int t = threadIdx.x;
  float* row = out + (size_t)blockIdx.x * N_TOK;

  float4 v[8];
  float m = -3.402823466e+38f;
  #pragma unroll
  for (int i = 0; i < 8; ++i) {
    v[i] = *(((const float4*)row) + t + 256 * i);
    m = fmaxf(m, fmaxf(fmaxf(v[i].x, v[i].y), fmaxf(v[i].z, v[i].w)));
  }
  red[t] = m; __syncthreads();
  for (int o = 128; o > 0; o >>= 1) {
    if (t < o) red[t] = fmaxf(red[t], red[t + o]);
    __syncthreads();
  }
  const float M = red[0];
  __syncthreads();

  float s = 0.0f;
  #pragma unroll
  for (int i = 0; i < 8; ++i) {
    v[i].x = __expf(v[i].x - M);
    v[i].y = __expf(v[i].y - M);
    v[i].z = __expf(v[i].z - M);
    v[i].w = __expf(v[i].w - M);
    s += v[i].x + v[i].y + v[i].z + v[i].w;
  }
  red[t] = s; __syncthreads();
  for (int o = 128; o > 0; o >>= 1) {
    if (t < o) red[t] += red[t + o];
    __syncthreads();
  }
  const float inv = 1.0f / red[0];

  #pragma unroll
  for (int i = 0; i < 8; ++i) {
    v[i].x *= inv; v[i].y *= inv; v[i].z *= inv; v[i].w *= inv;
    *(((float4*)row) + t + 256 * i) = v[i];
  }
}

extern "C" void kernel_launch(void* const* d_in, const int* in_sizes, int n_in,
                              void* d_out, int out_size, void* d_ws, size_t ws_size,
                              hipStream_t stream) {
  (void)in_sizes; (void)n_in; (void)d_ws; (void)ws_size; (void)out_size;
  const float* q   = (const float*)d_in[0];
  const float* kin = (const float*)d_in[1];
  float*       out = (float*)d_out;

  dim3 grid(N_TOK / 128, N_TOK / 128);       // 64 x 64 block tiles
  fused_gram_gemm<<<grid, 128, 0, stream>>>(q, kin, out);
  softmax_rows<<<N_TOK, 256, 0, stream>>>(out);
}
